// SyncNetClassifier_22548578304601
// MI455X (gfx1250) — compile-verified, matrix operand order checked by reading, not run on hardware
//
#include <hip/hip_runtime.h>
#include <cstdint>
#include <cstddef>

// ---------------------------------------------------------------------------
// Model dims (fixed by reference)
// ---------------------------------------------------------------------------
#define B_  256     // batch
#define S_  128     // encoder sequence length
#define H_  512     // hidden
#define T_  128     // decoder steps (== MAX_SEQ_LEN)
#define F_  128     // encoder input feature
#define SOS_ 128    // start token row in embedding table

#define KXMAX_ 1024           // max input width of any LSTM cell (decoder cell0)
#define PAD_   8              // LDS row padding (elements) -> 4-bank rotation/row

// ---------------------------------------------------------------------------
// bf16 helpers (raw u16 storage; conversions done explicitly)
// ---------------------------------------------------------------------------
typedef __attribute__((ext_vector_type(16))) __bf16         v16bf;
typedef __attribute__((ext_vector_type(8)))  float          v8f;
typedef __attribute__((ext_vector_type(8)))  unsigned short u16x8;

__device__ __forceinline__ float bf2f(unsigned short u) {
  return __uint_as_float(((unsigned int)u) << 16);
}
__device__ __forceinline__ unsigned short f2bf(float f) {
  unsigned int x = __float_as_uint(f);
  x += 0x7fffu + ((x >> 16) & 1u);          // round-to-nearest-even
  return (unsigned short)(x >> 16);
}
__device__ __forceinline__ float sigmf(float x) { return 1.0f / (1.0f + __expf(-x)); }

// ---------------------------------------------------------------------------
// WMMA fragment loaders.
// wave32 bf16 A (16x32) / B (32x16 from row-major W[N,K]) fragment:
//   lane L: row (or col) = base + (L & 15)
//   K elements: [k0 + (L>>4)*8 .. +7] then [k0 + 16 + (L>>4)*8 .. +7]
// Both chunks are 16B contiguous -> global_load_b128 / ds_load_b128 x2.
// ---------------------------------------------------------------------------
union FragU { v16bf bf; u16x8 h[2]; };

__device__ __forceinline__ v16bf load_frag(const unsigned short* __restrict__ base,
                                           int ld, int row, int k0, int lane) {
  const unsigned short* p = base + (size_t)row * ld + k0 + ((lane >> 4) << 3);
  FragU u;
  u.h[0] = *(const u16x8*)(p);
  u.h[1] = *(const u16x8*)(p + 16);
  return u.bf;
}

// A-fragment from LDS slab (row stride = ldpad elements, 16B aligned)
__device__ __forceinline__ v16bf load_frag_s(const unsigned short* base,
                                             int ldpad, int row, int k0, int lane) {
  const unsigned short* p = base + row * ldpad + k0 + ((lane >> 4) << 3);
  FragU u;
  u.h[0] = *(const u16x8*)(p);
  u.h[1] = *(const u16x8*)(p + 16);
  return u.bf;
}

__device__ __forceinline__ v8f wmma_bf16(v16bf a, v16bf b, v8f c) {
  return __builtin_amdgcn_wmma_f32_16x16x32_bf16(false, a, false, b, (short)0, c,
                                                 false, false);
}

// ---------------------------------------------------------------------------
// CDNA5 async Global->LDS copy (ASYNCcnt-tracked). Each lane moves 16B from
// its global address to its LDS byte offset.  Low 32 bits of a generic
// pointer to __shared__ == wave-relative LDS byte offset.
// ---------------------------------------------------------------------------
__device__ __forceinline__ void async_copy_b128(const unsigned short* g,
                                                const unsigned short* l) {
  unsigned lds = (unsigned)(size_t)l;
  unsigned long long ga = (unsigned long long)(size_t)g;
  asm volatile("global_load_async_to_lds_b128 %0, %1, off"
               :: "v"(lds), "v"(ga) : "memory");
}
__device__ __forceinline__ void wait_async0() {
  asm volatile("s_wait_asynccnt 0x0" ::: "memory");
}

// ---------------------------------------------------------------------------
// Fused LSTM cell:  z = X @ Wih^T + Hprev @ Whh^T + b ; gates i,f,g,o ; update
// Grid: 64 blocks x 256 threads = 512 waves; wave -> (mTile 0..15, hidTile 0..31).
// All 8 waves of a block share one batch tile -> the 16 A-rows of X and Hprev
// are staged once per block into LDS via async copies (A reuse = 8x), while
// B (weights, used exactly once per step) streams global->VGPR.
// Each wave holds the 4 gate tiles (i/f/g/o) of its 16x16 slab -> pointwise
// LSTM update fully in registers.  C fp32 [B,512] in place (element owned by
// exactly one wave).  H written as bf16.
// ---------------------------------------------------------------------------
__global__ void lstm_cell(const unsigned short* __restrict__ X, int ldx, int kx,
                          const unsigned short* __restrict__ Hp, int ldh,
                          const unsigned short* __restrict__ Wih,
                          const unsigned short* __restrict__ Whh,
                          const float* __restrict__ bias,
                          float* __restrict__ C,
                          unsigned short* __restrict__ Ho, int ldho,
                          unsigned short* __restrict__ Hcopy) {
  __shared__ __align__(16) unsigned short sx[16 * (KXMAX_ + PAD_)];
  __shared__ __align__(16) unsigned short sh[16 * (H_ + PAD_)];

  int wave = (blockIdx.x * blockDim.x + threadIdx.x) >> 5;
  int lane = threadIdx.x & 31;
  int mT = blockIdx.x >> 2;    // batch tile (same for all 8 waves of a block)
  int jT = wave & 31;          // hidden tile 0..31
  int m0 = mT * 16, n0 = jT * 16;
  int ldpx = kx + PAD_;

  // ---- async-stage the block's 16 A-rows of X and Hprev into LDS ----
  {
    int nckx = kx >> 3;                       // 16B chunks per X row
    for (int c = threadIdx.x; c < 16 * nckx; c += 256) {
      int row = c / nckx, k8 = (c - row * nckx) << 3;
      async_copy_b128(X + (size_t)(m0 + row) * ldx + k8, sx + row * ldpx + k8);
    }
    int nckh = H_ >> 3;                       // 16B chunks per H row
    for (int c = threadIdx.x; c < 16 * nckh; c += 256) {
      int row = c / nckh, k8 = (c - row * nckh) << 3;
      async_copy_b128(Hp + (size_t)(m0 + row) * ldh + k8, sh + row * (H_ + PAD_) + k8);
    }
    wait_async0();
    __syncthreads();
  }

  v8f z8 = {0.f, 0.f, 0.f, 0.f, 0.f, 0.f, 0.f, 0.f};
  v8f acc[4];
#pragma unroll
  for (int g = 0; g < 4; ++g) acc[g] = z8;

  int lrow = lane & 15;
  for (int k0 = 0; k0 < kx; k0 += 32) {
    v16bf a = load_frag_s(sx, ldpx, lrow, k0, lane);
#pragma unroll
    for (int g = 0; g < 4; ++g) {
      v16bf b = load_frag(Wih, kx, g * H_ + n0 + (lane & 15), k0, lane);
      acc[g] = wmma_bf16(a, b, acc[g]);
    }
  }
  for (int k0 = 0; k0 < H_; k0 += 32) {
    v16bf a = load_frag_s(sh, H_ + PAD_, lrow, k0, lane);
#pragma unroll
    for (int g = 0; g < 4; ++g) {
      v16bf b = load_frag(Whh, H_, g * H_ + n0 + (lane & 15), k0, lane);
      acc[g] = wmma_bf16(a, b, acc[g]);
    }
  }

  int n = n0 + (lane & 15);
  float bI = bias[n], bF = bias[H_ + n], bG = bias[2 * H_ + n], bO = bias[3 * H_ + n];
  int mh = m0 + ((lane >> 4) << 3);
#pragma unroll
  for (int j = 0; j < 8; ++j) {
    int m = mh + j;
    float zi = acc[0][j] + bI;
    float zf = acc[1][j] + bF;
    float zg = acc[2][j] + bG;
    float zo = acc[3][j] + bO;
    float c  = C[m * H_ + n];
    float cn = sigmf(zf) * c + sigmf(zi) * tanhf(zg);
    float hn = sigmf(zo) * tanhf(cn);
    C[m * H_ + n] = cn;
    unsigned short hb = f2bf(hn);
    Ho[(size_t)m * ldho + n] = hb;
    if (Hcopy) Hcopy[(size_t)m * H_ + n] = hb;
  }
}

// ---------------------------------------------------------------------------
// Generic WMMA GEMM:  out = act( A @ W^T + bias ),  A:[M,K] bf16 (lda),
// W:[N,K] bf16 row-major, out fp32 or bf16 (ldo).  Grid-stride over 16x16 tiles.
// ---------------------------------------------------------------------------
__global__ void gemm_wmma(const unsigned short* __restrict__ A, int lda,
                          const unsigned short* __restrict__ W, int K,
                          const float* __restrict__ bias,
                          int M, int N, int ldo,
                          float* __restrict__ outF,
                          unsigned short* __restrict__ outB, int relu) {
  int lane  = threadIdx.x & 31;
  int wave  = (blockIdx.x * blockDim.x + threadIdx.x) >> 5;
  int nwave = (gridDim.x * blockDim.x) >> 5;
  int ntN   = N >> 4;
  int tiles = (M >> 4) * ntN;
  for (int t = wave; t < tiles; t += nwave) {
    int mT = t / ntN, nT = t - mT * ntN;
    int m0 = mT * 16, n0 = nT * 16;
    v8f acc = {0.f, 0.f, 0.f, 0.f, 0.f, 0.f, 0.f, 0.f};
    int arow = m0 + (lane & 15);
    int brow = n0 + (lane & 15);
    for (int k0 = 0; k0 < K; k0 += 32) {
      v16bf a = load_frag(A, lda, arow, k0, lane);
      v16bf b = load_frag(W, K, brow, k0, lane);
      acc = wmma_bf16(a, b, acc);
    }
    int n = n0 + (lane & 15);
    float bb = bias ? bias[n] : 0.f;
    int mh = m0 + ((lane >> 4) << 3);
#pragma unroll
    for (int j = 0; j < 8; ++j) {
      float v = acc[j] + bb;
      if (relu) v = fmaxf(v, 0.f);
      size_t idx = (size_t)(mh + j) * ldo + n;
      if (outF) outF[idx] = v;
      else      outB[idx] = f2bf(v);
    }
  }
}

// ---------------------------------------------------------------------------
// Attention scores: scores[b,s] = sum_k tanh(ot[s,b,k] + ht[b,k]) * fc3[k]
// One block (64 threads) per (s,b); float4 coalesced loads + LDS tree reduce.
// ---------------------------------------------------------------------------
__global__ void attn_score(const float* __restrict__ ot,
                           const float* __restrict__ ht,
                           const float* __restrict__ fc3,
                           float* __restrict__ scores) {
  int sb = blockIdx.x;
  int s = sb >> 8, b = sb & 255;
  int tid = threadIdx.x;                       // 64 threads * float4 = 256
  const float4* po = (const float4*)(ot + ((size_t)s * B_ + b) * 256);
  const float4* ph = (const float4*)(ht + (size_t)b * 256);
  const float4* pf = (const float4*)fc3;
  float4 o = po[tid], h = ph[tid], f = pf[tid];
  float acc = tanhf(o.x + h.x) * f.x + tanhf(o.y + h.y) * f.y +
              tanhf(o.z + h.z) * f.z + tanhf(o.w + h.w) * f.w;
  __shared__ float red[64];
  red[tid] = acc; __syncthreads();
  for (int off = 32; off > 0; off >>= 1) {
    if (tid < off) red[tid] += red[tid + off];
    __syncthreads();
  }
  if (tid == 0) scores[b * S_ + s] = red[0];
}

// ---------------------------------------------------------------------------
// Softmax over s (per b) + context: ctx[b,:] = sum_s w[s] * enc_outs[s,b,:]
// ctx written (bf16) into xcat[:,512:] (this step's MLP) and xin[:,512:]
// (next step's cell0 input).  One block (128 threads) per batch element.
// ---------------------------------------------------------------------------
__global__ void attn_ctx(const float* __restrict__ scores,
                         const unsigned short* __restrict__ enc_outs,
                         unsigned short* __restrict__ xcat,
                         unsigned short* __restrict__ xin) {
  int b = blockIdx.x, tid = threadIdx.x;       // 128 threads, tid == s
  __shared__ float wls[S_];
  __shared__ float red[S_];
  float sc = scores[b * S_ + tid];
  red[tid] = sc; __syncthreads();
  for (int off = 64; off > 0; off >>= 1) {
    if (tid < off) red[tid] = fmaxf(red[tid], red[tid + off]);
    __syncthreads();
  }
  float mx = red[0]; __syncthreads();
  float e = __expf(sc - mx);
  wls[tid] = e; red[tid] = e; __syncthreads();
  for (int off = 64; off > 0; off >>= 1) {
    if (tid < off) red[tid] += red[tid + off];
    __syncthreads();
  }
  float inv = 1.0f / red[0];
  __syncthreads();
  for (int d = tid; d < H_; d += 128) {
    float acc = 0.f;
#pragma unroll 4
    for (int s = 0; s < S_; ++s)
      acc += wls[s] * bf2f(enc_outs[((size_t)s * B_ + b) * H_ + d]);
    unsigned short cv = f2bf(acc * inv);
    xcat[(size_t)b * 1024 + 512 + d] = cv;
    xin [(size_t)b * 1024 + 512 + d] = cv;
  }
}

// ---------------------------------------------------------------------------
// Head: out = hd2 @ w3^T + b3 ; logits -> d_out[t]; argmax (first-max
// tie-break, matching jnp.argmax); gather next token embedding into xin[:,:512].
// One block (128 threads) per batch element.
// ---------------------------------------------------------------------------
__global__ void head_argmax(const unsigned short* __restrict__ hd2,
                            const unsigned short* __restrict__ w3,
                            const float* __restrict__ b3,
                            const unsigned short* __restrict__ embb,
                            float* __restrict__ out_t,
                            unsigned short* __restrict__ xin) {
  int b = blockIdx.x, tid = threadIdx.x;       // 128 threads, tid == n
  __shared__ float xv[256];
  for (int k = tid; k < 256; k += 128) xv[k] = bf2f(hd2[(size_t)b * 256 + k]);
  __syncthreads();
  float acc = b3[tid];
  const u16x8* wr8 = (const u16x8*)(w3 + (size_t)tid * 256);
#pragma unroll 4
  for (int k8 = 0; k8 < 32; ++k8) {
    u16x8 w = wr8[k8];
#pragma unroll
    for (int j = 0; j < 8; ++j) acc += xv[k8 * 8 + j] * bf2f(w[j]);
  }
  out_t[(size_t)b * T_ + tid] = acc;
  __shared__ float mv[128];
  __shared__ int   mi[128];
  mv[tid] = acc; mi[tid] = tid; __syncthreads();
  for (int off = 64; off > 0; off >>= 1) {
    if (tid < off) {
      float ov = mv[tid + off]; int oi = mi[tid + off];
      if (ov > mv[tid] || (ov == mv[tid] && oi < mi[tid])) { mv[tid] = ov; mi[tid] = oi; }
    }
    __syncthreads();
  }
  int tok = mi[0];
  for (int k = tid; k < H_; k += 128)
    xin[(size_t)b * 1024 + k] = embb[(size_t)tok * H_ + k];
}

// ---------------------------------------------------------------------------
// Utility kernels
// ---------------------------------------------------------------------------
__global__ void cvt_f32_bf16(const float* __restrict__ s, unsigned short* __restrict__ d, int n) {
  for (int i = blockIdx.x * blockDim.x + threadIdx.x; i < n; i += gridDim.x * blockDim.x)
    d[i] = f2bf(s[i]);
}
__global__ void zero_f32(float* p, int n) {
  for (int i = blockIdx.x * blockDim.x + threadIdx.x; i < n; i += gridDim.x * blockDim.x)
    p[i] = 0.f;
}
__global__ void zero_bf(unsigned short* p, int n) {
  for (int i = blockIdx.x * blockDim.x + threadIdx.x; i < n; i += gridDim.x * blockDim.x)
    p[i] = 0;
}
// After encoder: xcat0[:, :512] = final h2 ; xin = [emb[SOS] | 0]
__global__ void dec_init(const unsigned short* __restrict__ h2fin,
                         const unsigned short* __restrict__ embb,
                         unsigned short* __restrict__ xcat0,
                         unsigned short* __restrict__ xin) {
  int i = blockIdx.x * blockDim.x + threadIdx.x;      // 0 .. B*512-1
  if (i >= B_ * H_) return;
  int b = i >> 9, k = i & 511;
  xcat0[(size_t)b * 1024 + k]       = h2fin[i];
  xin  [(size_t)b * 1024 + k]       = embb[SOS_ * H_ + k];
  xin  [(size_t)b * 1024 + 512 + k] = 0;
}

// ---------------------------------------------------------------------------
// Host driver: all launches on `stream` (graph-capture safe).
// ---------------------------------------------------------------------------
extern "C" void kernel_launch(void* const* d_in, const int* in_sizes, int n_in,
                              void* d_out, int out_size, void* d_ws, size_t ws_size,
                              hipStream_t stream) {
  (void)in_sizes; (void)n_in; (void)out_size; (void)ws_size;

  const float* eb0   = (const float*)d_in[3];
  const float* eb1   = (const float*)d_in[6];
  const float* eb2   = (const float*)d_in[9];
  const float* db0   = (const float*)d_in[12];
  const float* db1   = (const float*)d_in[15];
  const float* db2   = (const float*)d_in[18];
  const float* fc1b  = (const float*)d_in[21];
  const float* fc2b  = (const float*)d_in[23];
  const float* fc3w  = (const float*)d_in[24];
  const float* mb1   = (const float*)d_in[26];
  const float* mb2   = (const float*)d_in[28];
  const float* mb3   = (const float*)d_in[30];

  char* base = (char*)d_ws;
  size_t off = 0;
  auto alloc = [&](size_t bytes) -> char* {
    char* p = base + off;
    off += (bytes + 255) & ~(size_t)255;
    return p;
  };
  auto abf = [&](size_t n) { return (unsigned short*)alloc(n * 2); };
  auto af  = [&](size_t n) { return (float*)alloc(n * 4); };

  // bf16 weight / activation buffers
  unsigned short* xfeat  = abf((size_t)B_ * S_ * F_);
  unsigned short* wih0b  = abf(2048 * F_);
  unsigned short* whh0b  = abf(2048 * H_);
  unsigned short* wih1b  = abf(2048 * H_);
  unsigned short* whh1b  = abf(2048 * H_);
  unsigned short* wih2b  = abf(2048 * H_);
  unsigned short* whh2b  = abf(2048 * H_);
  unsigned short* dwih0b = abf(2048 * 1024);
  unsigned short* dwhh0b = abf(2048 * H_);
  unsigned short* dwih1b = abf(2048 * H_);
  unsigned short* dwhh1b = abf(2048 * H_);
  unsigned short* dwih2b = abf(2048 * H_);
  unsigned short* dwhh2b = abf(2048 * H_);
  unsigned short* embb   = abf(130 * H_);
  unsigned short* fc1wb  = abf(256 * H_);
  unsigned short* fc2wb  = abf(256 * H_);
  unsigned short* w1b    = abf(512 * 1024);
  unsigned short* w2b    = abf(256 * 512);
  unsigned short* w3b    = abf(128 * 256);
  unsigned short* enc_outs = abf((size_t)S_ * B_ * H_);
  unsigned short* hb[3][2];
  for (int l = 0; l < 3; ++l) { hb[l][0] = abf(B_ * H_); hb[l][1] = abf(B_ * H_); }
  unsigned short* xcat[2] = { abf(B_ * 1024), abf(B_ * 1024) };
  unsigned short* xin    = abf(B_ * 1024);
  unsigned short* m1     = abf(B_ * 512);
  unsigned short* m2     = abf(B_ * 256);
  // fp32 buffers
  float* cb[3] = { af(B_ * H_), af(B_ * H_), af(B_ * H_) };
  float* ot     = af((size_t)S_ * B_ * 256);
  float* ht     = af(B_ * 256);
  float* scores = af(B_ * S_);

  auto CVT = [&](int idx, unsigned short* dst, int n) {
    int g = (n + 255) / 256; if (g > 4096) g = 4096;
    cvt_f32_bf16<<<g, 256, 0, stream>>>((const float*)d_in[idx], dst, n);
  };

  // ---- Phase 0: convert fp32 -> bf16 (weights + features), init state ----
  CVT(0,  xfeat,  B_ * S_ * F_);
  CVT(1,  wih0b,  2048 * F_);   CVT(2,  whh0b,  2048 * H_);
  CVT(4,  wih1b,  2048 * H_);   CVT(5,  whh1b,  2048 * H_);
  CVT(7,  wih2b,  2048 * H_);   CVT(8,  whh2b,  2048 * H_);
  CVT(10, dwih0b, 2048 * 1024); CVT(11, dwhh0b, 2048 * H_);
  CVT(13, dwih1b, 2048 * H_);   CVT(14, dwhh1b, 2048 * H_);
  CVT(16, dwih2b, 2048 * H_);   CVT(17, dwhh2b, 2048 * H_);
  CVT(19, embb,   130 * H_);
  CVT(20, fc1wb,  256 * H_);    CVT(22, fc2wb,  256 * H_);
  CVT(25, w1b,    512 * 1024);  CVT(27, w2b,    256 * 512);
  CVT(29, w3b,    128 * 256);
  for (int l = 0; l < 3; ++l) {
    zero_f32<<<512, 256, 0, stream>>>(cb[l], B_ * H_);
    zero_bf <<<512, 256, 0, stream>>>(hb[l][0], B_ * H_);
  }

  // ---- Phase 1: encoder (3-layer LSTM, 128 steps), h ping-pong ----
  for (int t = 0; t < S_; ++t) {
    int cur = t & 1, nxt = cur ^ 1;
    lstm_cell<<<64, 256, 0, stream>>>(xfeat + (size_t)t * F_, S_ * F_, F_,
                                      hb[0][cur], H_, wih0b, whh0b, eb0, cb[0],
                                      hb[0][nxt], H_, nullptr);
    lstm_cell<<<64, 256, 0, stream>>>(hb[0][nxt], H_, H_,
                                      hb[1][cur], H_, wih1b, whh1b, eb1, cb[1],
                                      hb[1][nxt], H_, nullptr);
    lstm_cell<<<64, 256, 0, stream>>>(hb[1][nxt], H_, H_,
                                      hb[2][cur], H_, wih2b, whh2b, eb2, cb[2],
                                      hb[2][nxt], H_, enc_outs + (size_t)t * B_ * H_);
  }
  // final hidden states land in hb[l][0] (128 steps, even)

  // ---- Phase 2: ot = enc_outs @ fc2^T + fc2_b  (loop-invariant) ----
  gemm_wmma<<<512, 256, 0, stream>>>(enc_outs, H_, fc2wb, H_, fc2b,
                                     S_ * B_, 256, 256, ot, nullptr, 0);
  dec_init<<<(B_ * H_ + 255) / 256, 256, 0, stream>>>(hb[2][0], embb, xcat[0], xin);

  // ---- Phase 3: decoder (128 steps) ----
  float* out_f = (float*)d_out;
  for (int t = 0; t < T_; ++t) {
    int cur = t & 1, nxt = cur ^ 1;
    lstm_cell<<<64, 256, 0, stream>>>(xin, 1024, 1024,
                                      hb[0][cur], H_, dwih0b, dwhh0b, db0, cb[0],
                                      hb[0][nxt], H_, nullptr);
    lstm_cell<<<64, 256, 0, stream>>>(hb[0][nxt], H_, H_,
                                      hb[1][cur], H_, dwih1b, dwhh1b, db1, cb[1],
                                      hb[1][nxt], H_, nullptr);
    lstm_cell<<<64, 256, 0, stream>>>(hb[1][nxt], H_, H_,
                                      xcat[cur], 1024, dwih2b, dwhh2b, db2, cb[2],
                                      xcat[nxt], 1024, nullptr);
    // ht = h2 @ fc1^T + fc1_b
    gemm_wmma<<<32, 256, 0, stream>>>(xcat[nxt], 1024, fc1wb, H_, fc1b,
                                      B_, 256, 256, ht, nullptr, 0);
    attn_score<<<S_ * B_, 64, 0, stream>>>(ot, ht, fc3w, scores);
    attn_ctx<<<B_, 128, 0, stream>>>(scores, enc_outs, xcat[nxt], xin);
    // MLP on [h2 | new_ctx]
    gemm_wmma<<<64, 256, 0, stream>>>(xcat[nxt], 1024, w1b, 1024, mb1,
                                      B_, 512, 512, nullptr, m1, 1);
    gemm_wmma<<<32, 256, 0, stream>>>(m1, 512, w2b, 512, mb2,
                                      B_, 256, 256, nullptr, m2, 1);
    head_argmax<<<B_, 128, 0, stream>>>(m2, w3b, mb3, embb,
                                        out_f + (size_t)t * B_ * T_, xin);
  }
}
